// PaiNNRepresentation_10170482557553
// MI455X (gfx1250) — matrix operand order repaired and checked by reading, not compiled, not run on hardware
//
#include <hip/hip_runtime.h>
#include <hip/hip_bf16.h>

#define F 128
#define NRBF 20
#define PI_F 3.14159265358979f
#define CUTOFF_F 5.0f
#define EPS_F 1e-8f
#define FILT_STRIDE 1152   // N_LAYERS*3*F

typedef __bf16 bf16_t;
typedef bf16_t v2bf  __attribute__((ext_vector_type(2)));
typedef bf16_t v8bf  __attribute__((ext_vector_type(8)));
typedef bf16_t v16bf __attribute__((ext_vector_type(16)));
typedef float  v8f   __attribute__((ext_vector_type(8)));

__device__ __forceinline__ float silu_f(float x) {
  return x * __builtin_amdgcn_rcpf(1.0f + __expf(-x));  // v_rcp_f32, not IEEE divide
}

// ---------------------------------------------------------------------------
// q[a,f] = emb[elems[a], f]
// ---------------------------------------------------------------------------
__global__ void embed_kernel(const int* __restrict__ elems, const float* __restrict__ emb,
                             float* __restrict__ q, int nAtoms) {
  int i = blockIdx.x * blockDim.x + threadIdx.x;
  if (i < nAtoms * F) {
    int a = i >> 7, f = i & (F - 1);
    q[i] = emb[elems[a] * F + f];
  }
}

// ---------------------------------------------------------------------------
// Per-edge record: phi[k]*fcut (20), dir (3), fcut (1) => 24 floats/edge.
// ---------------------------------------------------------------------------
__global__ void edge_geom_kernel(const float* __restrict__ edge_vec,
                                 float* __restrict__ rec, int nEdges) {
  int e = blockIdx.x * blockDim.x + threadIdx.x;
  if (e >= nEdges) return;
  float vx = edge_vec[e * 3 + 0], vy = edge_vec[e * 3 + 1], vz = edge_vec[e * 3 + 2];
  float d = sqrtf(vx * vx + vy * vy + vz * vz);
  float inv = (d > 0.0f) ? 1.0f / d : 0.0f;
  float nrm = (d == 0.0f) ? 1.0f : d;
  float fc = (d <= CUTOFF_F) ? 0.5f * (1.0f + __cosf(PI_F * d / CUTOFF_F)) : 0.0f;
  float* r = rec + e * 24;
  float sc = fc / nrm;
#pragma unroll
  for (int k = 0; k < NRBF; ++k) {
    float fr = (float)(k + 1) * (PI_F / CUTOFF_F);
    r[k] = __sinf(d * fr) * sc;
  }
  r[20] = vx * inv; r[21] = vy * inv; r[22] = vz * inv; r[23] = fc;
}

// ---------------------------------------------------------------------------
// One-time weight transform: W[K][N] f32 -> WT[N][K] bf16 (B-frag = one
// contiguous 32B load per lane: frag elem e <-> K = laneGroup*16 + e).
// ---------------------------------------------------------------------------
__global__ void wt_convert_kernel(const float* __restrict__ W, bf16_t* __restrict__ WT,
                                  int K, int N) {
  int i = blockIdx.x * blockDim.x + threadIdx.x;
  if (i < K * N) {
    int n = i / K, k = i % K;
    WT[i] = (bf16_t)W[k * N + n];
  }
}

// ---------------------------------------------------------------------------
// GEMM 1: A f32 [M,K] staged to LDS as bf16; B direct from WT bf16 [N][K];
// silu epilogue; bf16 output. Block 256 thr = 8 waves (2Mx4N), block tile
// 64x128, wave tile 32x32 -> 4 WMMAs/wave/K-step. K%32==0, N%128==0.
// ---------------------------------------------------------------------------
__global__ void gemm_f32a_silu_bf16out(const float* __restrict__ A,
                                       const bf16_t* __restrict__ WT,
                                       const float* __restrict__ bias,
                                       bf16_t* __restrict__ C,
                                       int M, int K, int N) {
  __shared__ bf16_t Asl[64][32];
  int t = threadIdx.x;
  int w = t >> 5, lane = t & 31;
  int wm = w >> 2, wn = w & 3;
  int lg = lane >> 4, lr = lane & 15;
  int mbase = blockIdx.y * 64;
  int nbase = blockIdx.x * 128;
  v8f acc[2][2];
#pragma unroll
  for (int s = 0; s < 2; ++s)
#pragma unroll
    for (int u = 0; u < 2; ++u) acc[s][u] = (v8f){0.f,0.f,0.f,0.f,0.f,0.f,0.f,0.f};

  for (int kk = 0; kk < K; kk += 32) {
#pragma unroll
    for (int p = 0; p < 4; ++p) {             // A tile 64x32 as float2 pairs
      int idx2 = t + p * 256;
      int r = idx2 >> 4, c = (idx2 & 15) * 2;
      int gr = mbase + r; if (gr >= M) gr = M - 1;
      float2 fv = *(const float2*)(A + (long)gr * K + kk + c);
      v2bf pr; pr[0] = (bf16_t)fv.x; pr[1] = (bf16_t)fv.y;
      *(v2bf*)&Asl[r][c] = pr;
    }
    __syncthreads();

    v16bf bfrag[2];
#pragma unroll
    for (int u = 0; u < 2; ++u) {
      int col = nbase + wn * 32 + u * 16 + lr;
      bfrag[u] = *(const v16bf*)(WT + (size_t)col * K + kk + lg * 16);
    }
#pragma unroll
    for (int s = 0; s < 2; ++s) {
      const bf16_t* arow = &Asl[wm * 32 + s * 16 + lr][0];
      v16bf afrag = *(const v16bf*)(arow + lg * 16);
#pragma unroll
      for (int u = 0; u < 2; ++u)
        acc[s][u] = __builtin_amdgcn_wmma_f32_16x16x32_bf16(false, afrag, false, bfrag[u],
                                                            (short)0, acc[s][u], false, false);
    }
    __syncthreads();
  }

  if (mbase + 64 <= M) {  // uniform fast path: no per-element masking
#pragma unroll
    for (int s = 0; s < 2; ++s)
#pragma unroll
      for (int u = 0; u < 2; ++u) {
        int col = nbase + wn * 32 + u * 16 + lr;
        float bv = bias[col];
        int row0 = mbase + wm * 32 + s * 16 + lg * 8;
#pragma unroll
        for (int r = 0; r < 8; ++r)
          C[(long)(row0 + r) * N + col] = (bf16_t)silu_f(acc[s][u][r] + bv);
      }
  } else {
#pragma unroll
    for (int s = 0; s < 2; ++s)
#pragma unroll
      for (int u = 0; u < 2; ++u) {
        int col = nbase + wn * 32 + u * 16 + lr;
        float bv = bias[col];
#pragma unroll
        for (int r = 0; r < 8; ++r) {
          int row = mbase + wm * 32 + s * 16 + lg * 8 + r;
          if (row < M) C[(long)row * N + col] = (bf16_t)silu_f(acc[s][u][r] + bv);
        }
      }
  }
}

// ---------------------------------------------------------------------------
// GEMM 2: A already bf16 [M,K]; fragments loaded directly from global;
// zero LDS, zero barriers. f32 output, no activation.
// ---------------------------------------------------------------------------
__global__ void gemm_a16_f32out(const bf16_t* __restrict__ A,
                                const bf16_t* __restrict__ WT,
                                const float* __restrict__ bias,
                                float* __restrict__ C,
                                int M, int K, int N) {
  int t = threadIdx.x;
  int w = t >> 5, lane = t & 31;
  int wm = w >> 2, wn = w & 3;
  int lg = lane >> 4, lr = lane & 15;
  int mbase = blockIdx.y * 64;
  int nbase = blockIdx.x * 128;
  v8f acc[2][2];
#pragma unroll
  for (int s = 0; s < 2; ++s)
#pragma unroll
    for (int u = 0; u < 2; ++u) acc[s][u] = (v8f){0.f,0.f,0.f,0.f,0.f,0.f,0.f,0.f};

  int rowIdx[2];
#pragma unroll
  for (int s = 0; s < 2; ++s) {
    int rr = mbase + wm * 32 + s * 16 + lr;
    rowIdx[s] = (rr >= M) ? (M - 1) : rr;    // clamp; masked at store
  }

  for (int kk = 0; kk < K; kk += 32) {
    v16bf bfrag[2];
#pragma unroll
    for (int u = 0; u < 2; ++u) {
      int col = nbase + wn * 32 + u * 16 + lr;
      bfrag[u] = *(const v16bf*)(WT + (size_t)col * K + kk + lg * 16);
    }
#pragma unroll
    for (int s = 0; s < 2; ++s) {
      const bf16_t* arow = A + (size_t)rowIdx[s] * K + kk;
      v16bf afrag = *(const v16bf*)(arow + lg * 16);
#pragma unroll
      for (int u = 0; u < 2; ++u)
        acc[s][u] = __builtin_amdgcn_wmma_f32_16x16x32_bf16(false, afrag, false, bfrag[u],
                                                            (short)0, acc[s][u], false, false);
    }
  }

  if (mbase + 64 <= M) {  // uniform fast path
#pragma unroll
    for (int s = 0; s < 2; ++s)
#pragma unroll
      for (int u = 0; u < 2; ++u) {
        int col = nbase + wn * 32 + u * 16 + lr;
        float bv = bias[col];
        int row0 = mbase + wm * 32 + s * 16 + lg * 8;
#pragma unroll
        for (int r = 0; r < 8; ++r)
          C[(long)(row0 + r) * N + col] = acc[s][u][r] + bv;
      }
  } else {
#pragma unroll
    for (int s = 0; s < 2; ++s)
#pragma unroll
      for (int u = 0; u < 2; ++u) {
        int col = nbase + wn * 32 + u * 16 + lr;
        float bv = bias[col];
#pragma unroll
        for (int r = 0; r < 8; ++r) {
          int row = mbase + wm * 32 + s * 16 + lg * 8 + r;
          if (row < M) C[(long)row * N + col] = acc[s][u][r] + bv;
        }
      }
  }
}

// ---------------------------------------------------------------------------
// Edge message pass (roofline limiter). 32-bit index math throughout
// (max offset 9.6M elems). global_prefetch_b8 on next iteration's gathers.
// ---------------------------------------------------------------------------
__global__ void edge_message_kernel(const float* __restrict__ rec,
                                    const int* __restrict__ edge_index,
                                    const float* __restrict__ filt_w,
                                    const float* __restrict__ filt_b,
                                    const float* __restrict__ x,    // nAtoms x 384 f32
                                    const float* __restrict__ mu,   // nAtoms x 3 x 128
                                    float* __restrict__ q,          // atomic dst
                                    float* __restrict__ muacc,      // atomic dst
                                    int nEdges, int layer) {
  __shared__ float wl[NRBF][3 * F];
  __shared__ float bl[3 * F];
  __shared__ float recs[2][24];
  int t = threadIdx.x;
  for (int idx = t; idx < NRBF * 3 * F; idx += 256) {
    int r = idx / (3 * F), c = idx % (3 * F);
    wl[r][c] = filt_w[r * FILT_STRIDE + layer * 3 * F + c];
  }
  for (int c = t; c < 3 * F; c += 256) bl[c] = filt_b[layer * 3 * F + c];
  __syncthreads();

  int f = t & (F - 1), sub = t >> 7;
  int stride2 = (int)gridDim.x * 2;
  for (int base = (int)blockIdx.x * 2; base < nEdges; base += stride2) {
    if (t < 48) {
      int ee = base + (t / 24);
      if (ee < nEdges) recs[t / 24][t % 24] = rec[ee * 24 + (t % 24)];
    }
    __syncthreads();
    int e = base + sub;
    if (e < nEdges) {
      int en = base + stride2 + sub;          // prefetch next gathers
      if (en < nEdges) {
        int jn = edge_index[en];
        const float* xn = x + jn * 3 * F;
        const float* mn = mu + jn * 3 * F;
        __builtin_prefetch(xn + f, 0, 1);
        __builtin_prefetch(xn + F + f, 0, 1);
        __builtin_prefetch(xn + 2 * F + f, 0, 1);
        __builtin_prefetch(mn + f, 0, 1);
        __builtin_prefetch(mn + F + f, 0, 1);
        __builtin_prefetch(mn + 2 * F + f, 0, 1);
      }
      int j = edge_index[e];                  // idx_j
      int i = edge_index[nEdges + e];         // idx_i
      const float* R = recs[sub];
      float fc = R[23];
      float w0 = bl[f] * fc, w1 = bl[F + f] * fc, w2 = bl[2 * F + f] * fc;
#pragma unroll
      for (int r = 0; r < NRBF; ++r) {
        float p = R[r];
        w0 += p * wl[r][f];
        w1 += p * wl[r][F + f];
        w2 += p * wl[r][2 * F + f];
      }
      const float* xj = x + j * 3 * F;
      float xw0 = w0 * xj[f], xw1 = w1 * xj[F + f], xw2 = w2 * xj[2 * F + f];
      atomicAdd(&q[i * F + f], xw0);
      const float* muj = mu + j * 3 * F;
      float* mi = muacc + i * 3 * F;
#pragma unroll
      for (int d = 0; d < 3; ++d)
        atomicAdd(&mi[d * F + f], xw1 * R[20 + d] + xw2 * muj[d * F + f]);
    }
    __syncthreads();
  }
}

// ---------------------------------------------------------------------------
// Per-atom pre-update: mu += dmu; mu_Vn = ||mu_V||; ctx = [q, mu_Vn]
// ---------------------------------------------------------------------------
__global__ void update_pre_kernel(const float* __restrict__ q, float* __restrict__ mu,
                                  const float* __restrict__ muacc,
                                  const float* __restrict__ mmw,   // 128 x 256 slice
                                  float* __restrict__ ctx) {       // nAtoms x 256
  __shared__ float muL[3][F];
  int a = blockIdx.x, f = threadIdx.x;
#pragma unroll
  for (int d = 0; d < 3; ++d) {
    int o = (a * 3 + d) * F + f;
    float m = mu[o] + muacc[o];
    mu[o] = m;
    muL[d][f] = m;
  }
  __syncthreads();
  float v0 = 0.f, v1 = 0.f, v2 = 0.f;
  for (int k = 0; k < F; ++k) {
    float wv = mmw[k * 2 * F + f];
    v0 += muL[0][k] * wv; v1 += muL[1][k] * wv; v2 += muL[2][k] * wv;
  }
  float vn = sqrtf(v0 * v0 + v1 * v1 + v2 * v2 + EPS_F);
  ctx[a * 2 * F + f]     = q[a * F + f];
  ctx[a * 2 * F + F + f] = vn;
}

// ---------------------------------------------------------------------------
// Per-atom post-update: recompute mu_V/mu_W; q += dq + dqmu*<mu_V,mu_W>;
// mu += dmu*mu_W.
// ---------------------------------------------------------------------------
__global__ void update_post_kernel(float* __restrict__ q, float* __restrict__ mu,
                                   const float* __restrict__ mmw,
                                   const float* __restrict__ x2) {
  __shared__ float muL[3][F];
  int a = blockIdx.x, f = threadIdx.x;
#pragma unroll
  for (int d = 0; d < 3; ++d) muL[d][f] = mu[(a * 3 + d) * F + f];
  __syncthreads();
  float v0 = 0.f, v1 = 0.f, v2 = 0.f, w0 = 0.f, w1 = 0.f, w2 = 0.f;
  for (int k = 0; k < F; ++k) {
    float m0 = muL[0][k], m1 = muL[1][k], m2 = muL[2][k];
    float wv = mmw[k * 2 * F + f];
    float ww = mmw[k * 2 * F + F + f];
    v0 += m0 * wv; v1 += m1 * wv; v2 += m2 * wv;
    w0 += m0 * ww; w1 += m1 * ww; w2 += m2 * ww;
  }
  float s = v0 * w0 + v1 * w1 + v2 * w2;
  const float* xr = x2 + a * 3 * F;
  float dq = xr[f], dmu = xr[F + f], dqmu = xr[2 * F + f];
  q[a * F + f] += dq + dqmu * s;
  mu[(a * 3 + 0) * F + f] += dmu * w0;
  mu[(a * 3 + 1) * F + f] += dmu * w1;
  mu[(a * 3 + 2) * F + f] += dmu * w2;
}

// ---------------------------------------------------------------------------
extern "C" void kernel_launch(void* const* d_in, const int* in_sizes, int n_in,
                              void* d_out, int out_size, void* d_ws, size_t ws_size,
                              hipStream_t stream) {
  const int*   elems  = (const int*)d_in[0];
  const int*   eidx   = (const int*)d_in[1];
  const float* evec   = (const float*)d_in[2];
  const float* emb    = (const float*)d_in[3];
  const float* filt_w = (const float*)d_in[4];
  const float* filt_b = (const float*)d_in[5];
  const float* iw1    = (const float*)d_in[6];
  const float* ib1    = (const float*)d_in[7];
  const float* iw2    = (const float*)d_in[8];
  const float* ib2    = (const float*)d_in[9];
  const float* mmw    = (const float*)d_in[10];
  const float* mw1    = (const float*)d_in[11];
  const float* mb1    = (const float*)d_in[12];
  const float* mw2    = (const float*)d_in[13];
  const float* mb2    = (const float*)d_in[14];

  int nAtoms = in_sizes[0];
  int nEdges = in_sizes[2] / 3;

  float* q  = (float*)d_out;                 // nAtoms x 128
  float* mu = q + (long)nAtoms * F;          // nAtoms x 3 x 128

  float*  ws    = (float*)d_ws;
  float*  REC   = ws;                                  // nEdges*24 f32
  float*  XBUF  = REC   + (long)nEdges * 24;           // nAtoms*384 f32 (x / ctx)
  float*  X2BUF = XBUF  + (long)nAtoms * 3 * F;        // nAtoms*384 f32
  float*  MUACC = X2BUF + (long)nAtoms * 3 * F;        // nAtoms*384 f32
  bf16_t* HB    = (bf16_t*)(MUACC + (long)nAtoms * 3 * F); // nAtoms*128 bf16
  bf16_t* WT    = HB + (size_t)nAtoms * F;             // 12 transposed bf16 weights

  const long WT_LAYER = 147456;  // 16384 + 49152 + 32768 + 49152
  const long O_IW1 = 0, O_IW2 = 16384, O_MW1 = 65536, O_MW2 = 98304;

  for (int l = 0; l < 3; ++l) {
    bf16_t* wtl = WT + l * WT_LAYER;
    wt_convert_kernel<<<(F * F + 255) / 256, 256, 0, stream>>>(
        iw1 + (long)l * F * F, wtl + O_IW1, F, F);
    wt_convert_kernel<<<(F * 3 * F + 255) / 256, 256, 0, stream>>>(
        iw2 + (long)l * F * 3 * F, wtl + O_IW2, F, 3 * F);
    wt_convert_kernel<<<(2 * F * F + 255) / 256, 256, 0, stream>>>(
        mw1 + (long)l * 2 * F * F, wtl + O_MW1, 2 * F, F);
    wt_convert_kernel<<<(F * 3 * F + 255) / 256, 256, 0, stream>>>(
        mw2 + (long)l * F * 3 * F, wtl + O_MW2, F, 3 * F);
  }

  embed_kernel<<<(nAtoms * F + 255) / 256, 256, 0, stream>>>(elems, emb, q, nAtoms);
  hipMemsetAsync(mu, 0, (size_t)nAtoms * 3 * F * sizeof(float), stream);
  edge_geom_kernel<<<(nEdges + 255) / 256, 256, 0, stream>>>(evec, REC, nEdges);

  dim3 blk(256);
  int mtiles = (nAtoms + 63) / 64;
  for (int l = 0; l < 3; ++l) {
    bf16_t* wtl = WT + l * WT_LAYER;
    hipMemsetAsync(MUACC, 0, (size_t)nAtoms * 3 * F * sizeof(float), stream);

    gemm_f32a_silu_bf16out<<<dim3(F / 128, mtiles), blk, 0, stream>>>(
        q, wtl + O_IW1, ib1 + l * F, HB, nAtoms, F, F);
    gemm_a16_f32out<<<dim3(3 * F / 128, mtiles), blk, 0, stream>>>(
        HB, wtl + O_IW2, ib2 + l * 3 * F, XBUF, nAtoms, F, 3 * F);

    edge_message_kernel<<<2048, 256, 0, stream>>>(REC, eidx, filt_w, filt_b,
                                                  XBUF, mu, q, MUACC, nEdges, l);

    update_pre_kernel<<<nAtoms, F, 0, stream>>>(q, mu, MUACC,
                                                mmw + (long)l * F * 2 * F, XBUF);

    gemm_f32a_silu_bf16out<<<dim3(F / 128, mtiles), blk, 0, stream>>>(
        XBUF, wtl + O_MW1, mb1 + l * F, HB, nAtoms, 2 * F, F);
    gemm_a16_f32out<<<dim3(3 * F / 128, mtiles), blk, 0, stream>>>(
        HB, wtl + O_MW2, mb2 + l * 3 * F, X2BUF, nAtoms, F, 3 * F);

    update_post_kernel<<<nAtoms, F, 0, stream>>>(q, mu, mmw + (long)l * F * 2 * F, X2BUF);
  }
}
